// GroupQueryAttention_71983651881306
// MI455X (gfx1250) — compile-verified, matrix operand order checked by reading, not run on hardware
//
#include <hip/hip_runtime.h>
#include <hip/hip_bf16.h>
#include <stdint.h>

// ---------------------------------------------------------------- types ----
typedef __bf16 bf16_t;
typedef __attribute__((ext_vector_type(16))) __bf16        v16bf;
typedef __attribute__((ext_vector_type(8)))  float         v8f;
typedef __attribute__((ext_vector_type(4)))  unsigned int  u32x4;

#define HIDDEN   2048
#define KVDIM    512
#define N_HEADS  32
#define KVHEADS  8
#define HD       64
#define SEQ      2048
#define BATCH    2
#define NTOK     (BATCH * SEQ)   // 4096

// -------------------------------------------------- async-to-LDS support ---
#if defined(__has_builtin)
#  if __has_builtin(__builtin_amdgcn_global_load_async_to_lds_b128)
#    define HAVE_ASYNC 1
#  endif
#endif
#ifndef HAVE_ASYNC
#  define HAVE_ASYNC 0
#endif

#if HAVE_ASYNC
typedef int v4i __attribute__((vector_size(16)));
typedef __attribute__((address_space(1))) v4i glb_v4i;
typedef __attribute__((address_space(3))) v4i lds_v4i;
static __device__ __forceinline__ void async_cp16(const bf16_t* g, bf16_t* l) {
    // 16-byte global -> LDS async copy (ASYNCcnt tracked)
    __builtin_amdgcn_global_load_async_to_lds_b128(
        (glb_v4i*)(uintptr_t)g,   // global AS: same address value as flat
        (lds_v4i*)l,              // flat -> LDS addrspacecast (32-bit offset)
        0, 0);
}
template <int N>
static __device__ __forceinline__ void async_wait() {
#  if __has_builtin(__builtin_amdgcn_s_wait_asynccnt)
    __builtin_amdgcn_s_wait_asynccnt(N);
#  else
    asm volatile("s_wait_asynccnt %0" :: "i"(N) : "memory");
#  endif
}
#endif

// ---------------------------------------------------------------- utils ----
static __device__ __forceinline__ bf16_t bits2bf(unsigned short x) {
    union { unsigned short s; bf16_t b; } u; u.s = x; return u.b;
}
static __device__ __forceinline__ bf16_t f2bf(float f) {
    union { float f; unsigned u; } a; a.f = f;
    unsigned r = a.u + 0x7fffu + ((a.u >> 16) & 1u);   // round-to-nearest-even
    return bits2bf((unsigned short)(r >> 16));
}

// f32 -> bf16 grid-stride converter
__global__ void cvt_f32_bf16(const float* __restrict__ s, bf16_t* __restrict__ d, int n) {
    int i = blockIdx.x * blockDim.x + threadIdx.x;
    int stride = gridDim.x * blockDim.x;
    for (; i < n; i += stride) d[i] = f2bf(s[i]);
}

// Load one 16x32 bf16 A-style WMMA fragment from an LDS tile (ISA 7.12.2).
__device__ __forceinline__ v16bf ld_frag(const bf16_t* tile, int lda, int lane, int kchunk) {
    int r  = lane & 15;
    int ko = kchunk + ((lane >> 4) << 3);
    union { v16bf v; u32x4 u[2]; } f;
    f.u[0] = *(const u32x4*)(tile + r * lda + ko);
    f.u[1] = *(const u32x4*)(tile + r * lda + ko + 16);
    return f.v;
}

// ---------------------------------------------------------------- GEMM -----
// D[M,N] = A[M,K] * Wt[N,K]^T + bias[N]   (A, Wt K-contiguous bf16)
// 256 thr = 8 wave32 (4m x 2n). Block tile 128x128, K-step 32.
// HAVE_ASYNC: triple-buffered async-to-LDS pipeline, s_wait_asynccnt<=4.
#define LDA_PAD 40
#if HAVE_ASYNC
#define GBUF 3
#else
#define GBUF 1
#endif
template <bool OUTF32>
__global__ __launch_bounds__(256)
void gemm_bf16(const bf16_t* __restrict__ A, const bf16_t* __restrict__ Wt,
               const float* __restrict__ bias, void* __restrict__ outp,
               int M, int N, int K) {
    __shared__ bf16_t lA[GBUF][128 * LDA_PAD];
    __shared__ bf16_t lB[GBUF][128 * LDA_PAD];

    const int t = threadIdx.x, lane = t & 31, w = t >> 5;
    const int wm = w >> 1, wn = w & 1;
    const int bm = blockIdx.y * 128, bn = blockIdx.x * 128;

    const int r0 = t >> 2, c0 = (t & 3) << 3;
    const int sO0 = r0 * LDA_PAD + c0;
    const int sO1 = (r0 + 64) * LDA_PAD + c0;
    const bf16_t* gA0 = A  + (size_t)(bm + r0) * K + c0;
    const bf16_t* gA1 = A  + (size_t)(bm + r0 + 64) * K + c0;
    const bf16_t* gB0 = Wt + (size_t)(bn + r0) * K + c0;
    const bf16_t* gB1 = Wt + (size_t)(bn + r0 + 64) * K + c0;

    v8f c[2][4];
#pragma unroll
    for (int i = 0; i < 2; ++i)
#pragma unroll
        for (int j = 0; j < 4; ++j) c[i][j] = (v8f)0.0f;

#if HAVE_ASYNC
    // ---------------- async triple-buffered pipeline -----------------------
    async_cp16(gA0, &lA[0][sO0]); async_cp16(gA1, &lA[0][sO1]);
    async_cp16(gB0, &lB[0][sO0]); async_cp16(gB1, &lB[0][sO1]);
    gA0 += 32; gA1 += 32; gB0 += 32; gB1 += 32;
    if (K > 32) {
        async_cp16(gA0, &lA[1][sO0]); async_cp16(gA1, &lA[1][sO1]);
        async_cp16(gB0, &lB[1][sO0]); async_cp16(gB1, &lB[1][sO1]);
        gA0 += 32; gA1 += 32; gB0 += 32; gB1 += 32;
    }
    int cur = 0, tgt = 2;
    for (int k0 = 0; k0 < K; k0 += 32) {
        async_wait<4>();    // oldest tile (cur) landed; tile cur+1 may still fly
        __syncthreads();
        if (k0 + 64 < K) {
            async_cp16(gA0, &lA[tgt][sO0]); async_cp16(gA1, &lA[tgt][sO1]);
            async_cp16(gB0, &lB[tgt][sO0]); async_cp16(gB1, &lB[tgt][sO1]);
            gA0 += 32; gA1 += 32; gB0 += 32; gB1 += 32;
            tgt = (tgt == 2) ? 0 : tgt + 1;
        }
        v16bf a[2], b[4];
#pragma unroll
        for (int it = 0; it < 2; ++it)
            a[it] = ld_frag(&lA[cur][(wm * 32 + it * 16) * LDA_PAD], LDA_PAD, lane, 0);
#pragma unroll
        for (int jt = 0; jt < 4; ++jt)
            b[jt] = ld_frag(&lB[cur][(wn * 64 + jt * 16) * LDA_PAD], LDA_PAD, lane, 0);
#pragma unroll
        for (int it = 0; it < 2; ++it)
#pragma unroll
            for (int jt = 0; jt < 4; ++jt)
                c[it][jt] = __builtin_amdgcn_wmma_f32_16x16x32_bf16(
                    false, a[it], false, b[jt], (short)0, c[it][jt], false, false);
        cur = (cur == 2) ? 0 : cur + 1;
    }
#else
    // ---------------- register-staged pipeline (single LDS buffer) ---------
    u32x4 ra0 = *(const u32x4*)gA0, ra1 = *(const u32x4*)gA1;
    u32x4 rb0 = *(const u32x4*)gB0, rb1 = *(const u32x4*)gB1;
    gA0 += 32; gA1 += 32; gB0 += 32; gB1 += 32;
    for (int k0 = 0; k0 < K; k0 += 32) {
        __syncthreads();
        *(u32x4*)&lA[0][sO0] = ra0;  *(u32x4*)&lA[0][sO1] = ra1;
        *(u32x4*)&lB[0][sO0] = rb0;  *(u32x4*)&lB[0][sO1] = rb1;
        if (k0 + 32 < K) {
            ra0 = *(const u32x4*)gA0; ra1 = *(const u32x4*)gA1;
            rb0 = *(const u32x4*)gB0; rb1 = *(const u32x4*)gB1;
            gA0 += 32; gA1 += 32; gB0 += 32; gB1 += 32;
        }
        __syncthreads();
        v16bf a[2], b[4];
#pragma unroll
        for (int it = 0; it < 2; ++it)
            a[it] = ld_frag(&lA[0][(wm * 32 + it * 16) * LDA_PAD], LDA_PAD, lane, 0);
#pragma unroll
        for (int jt = 0; jt < 4; ++jt)
            b[jt] = ld_frag(&lB[0][(wn * 64 + jt * 16) * LDA_PAD], LDA_PAD, lane, 0);
#pragma unroll
        for (int it = 0; it < 2; ++it)
#pragma unroll
            for (int jt = 0; jt < 4; ++jt)
                c[it][jt] = __builtin_amdgcn_wmma_f32_16x16x32_bf16(
                    false, a[it], false, b[jt], (short)0, c[it][jt], false, false);
    }
#endif

    // epilogue
    const int rl = (lane >> 4) << 3;
    const int cl = lane & 15;
#pragma unroll
    for (int jt = 0; jt < 4; ++jt) {
        const int col = bn + wn * 64 + jt * 16 + cl;
        const float bs = bias[col];
#pragma unroll
        for (int it = 0; it < 2; ++it) {
            const int row0 = bm + wm * 32 + it * 16 + rl;
            if (OUTF32) {
                float* out = (float*)outp;
#pragma unroll
                for (int i = 0; i < 8; ++i)
                    out[(size_t)(row0 + i) * N + col] = c[it][jt][i] + bs;
            } else {
                bf16_t* out = (bf16_t*)outp;
#pragma unroll
                for (int i = 0; i < 8; ++i)
                    out[(size_t)(row0 + i) * N + col] = f2bf(c[it][jt][i] + bs);
            }
        }
    }
}

// ----------------------------------------------------- flash attention -----
// Grid: (SEQ/64, BATCH*N_HEADS). Block: 128 thr = 4 waves, wave owns 16 rows.
#define LDS_S 72
#if HAVE_ASYNC
#define KBUF 2
#else
#define KBUF 1
#endif
__global__ __launch_bounds__(128)
void gqa_flash(const bf16_t* __restrict__ Q, const bf16_t* __restrict__ Kc,
               const bf16_t* __restrict__ V, bf16_t* __restrict__ ctx) {
    __shared__ bf16_t lQ [64 * LDS_S];
    __shared__ bf16_t lK [KBUF][64 * LDS_S];
    __shared__ bf16_t lVt[64 * LDS_S];   // transposed: [d][key]
    __shared__ bf16_t lP [64 * LDS_S];

    const int t = threadIdx.x, lane = t & 31, w = t >> 5;
    const int bh = blockIdx.y;
    const int b  = bh >> 5, h = bh & 31, kh = h >> 2;   // GQA group = 4
    const size_t qrow0 = (size_t)b * SEQ + blockIdx.x * 64;
    const size_t krow0 = (size_t)b * SEQ;
    const int qcol = h * HD, kcol = kh * HD;

    const int ar = t >> 3;          // 0..15
    const int ac = (t & 7) << 3;    // 0..56

    // ---- load Q tile ----
    {
        u32x4 qreg[4];
#pragma unroll
        for (int u = 0; u < 4; ++u)
            qreg[u] = *(const u32x4*)(Q + (qrow0 + ar + 16 * u) * HIDDEN + qcol + ac);
#pragma unroll
        for (int u = 0; u < 4; ++u)
            *(u32x4*)(lQ + (ar + 16 * u) * LDS_S + ac) = qreg[u];
    }
    __syncthreads();

    v16bf aq[2];
    aq[0] = ld_frag(lQ + (w * 16) * LDS_S, LDS_S, lane, 0);
    aq[1] = ld_frag(lQ + (w * 16) * LDS_S, LDS_S, lane, 32);

    // constant all-ones B fragment (for row sums of P via WMMA)
    v16bf vones;
    {
        union { v16bf v; u32x4 u[2]; } oU;
        oU.u[0] = (u32x4)(0x3F803F80u);   // two bf16 1.0 per dword
        oU.u[1] = (u32x4)(0x3F803F80u);
        vones = oU.v;
    }

    float mrun[8], lrun[8];
    v8f o[4];
#pragma unroll
    for (int i = 0; i < 8; ++i) { mrun[i] = -1e30f; lrun[i] = 0.0f; }
#pragma unroll
    for (int nd = 0; nd < 4; ++nd) o[nd] = (v8f)0.0f;

    const int rl = (lane >> 4) << 3;
    const int cl = lane & 15;

    u32x4 vreg[4];
    auto load_v = [&](int kb) {
#pragma unroll
        for (int u = 0; u < 4; ++u)
            vreg[u] = *(const u32x4*)(V + (krow0 + kb + ar + 16 * u) * KVDIM + kcol + ac);
    };
#if HAVE_ASYNC
    auto issue_k = [&](int kb, int bufi) {
#pragma unroll
        for (int u = 0; u < 4; ++u)
            async_cp16(Kc + (krow0 + kb + ar + 16 * u) * KVDIM + kcol + ac,
                       &lK[bufi][(ar + 16 * u) * LDS_S + ac]);
    };
    issue_k(0, 0);
#else
    u32x4 kreg[4];
    auto load_k = [&](int kb) {
#pragma unroll
        for (int u = 0; u < 4; ++u)
            kreg[u] = *(const u32x4*)(Kc + (krow0 + kb + ar + 16 * u) * KVDIM + kcol + ac);
    };
    load_k(0);
#endif
    load_v(0);

    int kit = 0;
    for (int kb = 0; kb < SEQ; kb += 64, ++kit) {
#if HAVE_ASYNC
        async_wait<0>();   // K tile kb landed in lK[kit&1]
#endif
        __syncthreads();   // prior readers of lK/lVt done; K tile visible
        // ---- store transposed V tile ----
#pragma unroll
        for (int u = 0; u < 4; ++u) {
            int row = ar + 16 * u;
            unsigned vals[4] = {vreg[u].x, vreg[u].y, vreg[u].z, vreg[u].w};
#if !HAVE_ASYNC
            *(u32x4*)&lK[0][row * LDS_S + ac] = kreg[u];
#endif
#pragma unroll
            for (int e = 0; e < 4; ++e) {
                lVt[(ac + 2 * e)     * LDS_S + row] = bits2bf((unsigned short)(vals[e] & 0xffffu));
                lVt[(ac + 2 * e + 1) * LDS_S + row] = bits2bf((unsigned short)(vals[e] >> 16));
            }
        }
        if (kb + 64 < SEQ) {
#if HAVE_ASYNC
            issue_k(kb + 64, (kit + 1) & 1);
#else
            load_k(kb + 64);
#endif
            load_v(kb + 64);
        }
        __syncthreads();
        const bf16_t* lKcur = lK[(KBUF == 2) ? (kit & 1) : 0];

        // ---- S = Q * K^T ----
        v8f s[4];
#pragma unroll
        for (int jt = 0; jt < 4; ++jt) s[jt] = (v8f)0.0f;
#pragma unroll
        for (int jt = 0; jt < 4; ++jt)
#pragma unroll
            for (int kc = 0; kc < 2; ++kc) {
                v16bf bk = ld_frag(lKcur + (jt * 16) * LDS_S, LDS_S, lane, kc * 32);
                s[jt] = __builtin_amdgcn_wmma_f32_16x16x32_bf16(
                    false, aq[kc], false, bk, (short)0, s[jt], false, false);
            }

        // ---- online softmax: row max via 16-lane shuffles, fully unrolled ----
        float corr[8];
#pragma unroll
        for (int i = 0; i < 8; ++i) {
            const float sv0 = s[0][i] * 0.125f, sv1 = s[1][i] * 0.125f;
            const float sv2 = s[2][i] * 0.125f, sv3 = s[3][i] * 0.125f;
            float mb = fmaxf(fmaxf(sv0, sv1), fmaxf(sv2, sv3));
#pragma unroll
            for (int msk = 8; msk >= 1; msk >>= 1)
                mb = fmaxf(mb, __shfl_xor(mb, msk, 16));
            const float mn = fmaxf(mrun[i], mb);
            corr[i] = __expf(mrun[i] - mn);
            mrun[i] = mn;
            s[0][i] = __expf(sv0 - mn);
            s[1][i] = __expf(sv1 - mn);
            s[2][i] = __expf(sv2 - mn);
            s[3][i] = __expf(sv3 - mn);
#pragma unroll
            for (int nd = 0; nd < 4; ++nd) o[nd][i] *= corr[i];
        }

        // ---- P: C-layout -> bf16 LDS (wave-private rows) ----
#pragma unroll
        for (int j = 0; j < 4; ++j)
#pragma unroll
            for (int i = 0; i < 8; ++i)
                lP[(w * 16 + rl + i) * LDS_S + j * 16 + cl] = f2bf(s[j][i]);

        // ---- row sums via WMMA (P x ones): every column = rowsum ----
        v16bf ap[2];
        ap[0] = ld_frag(lP + (w * 16) * LDS_S, LDS_S, lane, 0);
        ap[1] = ld_frag(lP + (w * 16) * LDS_S, LDS_S, lane, 32);
        v8f lsv = (v8f)0.0f;
        lsv = __builtin_amdgcn_wmma_f32_16x16x32_bf16(
            false, ap[0], false, vones, (short)0, lsv, false, false);
        lsv = __builtin_amdgcn_wmma_f32_16x16x32_bf16(
            false, ap[1], false, vones, (short)0, lsv, false, false);
#pragma unroll
        for (int i = 0; i < 8; ++i) lrun[i] = lrun[i] * corr[i] + lsv[i];

        // ---- O += P * V ----
#pragma unroll
        for (int nd = 0; nd < 4; ++nd)
#pragma unroll
            for (int kc = 0; kc < 2; ++kc) {
                v16bf bv = ld_frag(lVt + (nd * 16) * LDS_S, LDS_S, lane, kc * 32);
                o[nd] = __builtin_amdgcn_wmma_f32_16x16x32_bf16(
                    false, ap[kc], false, bv, (short)0, o[nd], false, false);
            }
    }

    // ---- finalize ----
    float rinv[8];
#pragma unroll
    for (int i = 0; i < 8; ++i) rinv[i] = 1.0f / lrun[i];
#pragma unroll
    for (int nd = 0; nd < 4; ++nd)
#pragma unroll
        for (int i = 0; i < 8; ++i) {
            size_t row = qrow0 + w * 16 + rl + i;
            int    col = qcol + nd * 16 + cl;
            ctx[row * HIDDEN + col] = f2bf(o[nd][i] * rinv[i]);
        }
}

// ---------------------------------------------------------------- host -----
extern "C" void kernel_launch(void* const* d_in, const int* in_sizes, int n_in,
                              void* d_out, int out_size, void* d_ws, size_t ws_size,
                              hipStream_t stream) {
    (void)in_sizes; (void)n_in; (void)out_size;
    const float* x  = (const float*)d_in[0];
    const float* wq = (const float*)d_in[1];
    const float* bq = (const float*)d_in[2];
    const float* wk = (const float*)d_in[3];
    const float* bk = (const float*)d_in[4];
    const float* wv = (const float*)d_in[5];
    const float* bv = (const float*)d_in[6];
    const float* wo = (const float*)d_in[7];
    const float* bo = (const float*)d_in[8];

    uint8_t* ws = (uint8_t*)d_ws;
    size_t off = 0;
    auto carve = [&](size_t nelem) -> bf16_t* {
        bf16_t* p = (bf16_t*)(ws + off);
        off += (nelem * sizeof(bf16_t) + 255) & ~(size_t)255;
        return p;
    };
    bf16_t* xb  = carve((size_t)NTOK * HIDDEN);
    bf16_t* wqb = carve((size_t)HIDDEN * HIDDEN);
    bf16_t* wkb = carve((size_t)KVDIM * HIDDEN);
    bf16_t* wvb = carve((size_t)KVDIM * HIDDEN);
    bf16_t* wob = carve((size_t)HIDDEN * HIDDEN);
    bf16_t* Qb  = carve((size_t)NTOK * HIDDEN);
    bf16_t* Kb  = carve((size_t)NTOK * KVDIM);
    bf16_t* Vb  = carve((size_t)NTOK * KVDIM);
    bf16_t* Cb  = carve((size_t)NTOK * HIDDEN);
    if (off > ws_size) return;

    cvt_f32_bf16<<<2048, 256, 0, stream>>>(x,  xb,  NTOK * HIDDEN);
    cvt_f32_bf16<<<2048, 256, 0, stream>>>(wq, wqb, HIDDEN * HIDDEN);
    cvt_f32_bf16<<<1024, 256, 0, stream>>>(wk, wkb, KVDIM * HIDDEN);
    cvt_f32_bf16<<<1024, 256, 0, stream>>>(wv, wvb, KVDIM * HIDDEN);
    cvt_f32_bf16<<<2048, 256, 0, stream>>>(wo, wob, HIDDEN * HIDDEN);

    gemm_bf16<false><<<dim3(HIDDEN / 128, NTOK / 128), 256, 0, stream>>>(
        xb, wqb, bq, (void*)Qb, NTOK, HIDDEN, HIDDEN);
    gemm_bf16<false><<<dim3(KVDIM / 128, NTOK / 128), 256, 0, stream>>>(
        xb, wkb, bk, (void*)Kb, NTOK, KVDIM, HIDDEN);
    gemm_bf16<false><<<dim3(KVDIM / 128, NTOK / 128), 256, 0, stream>>>(
        xb, wvb, bv, (void*)Vb, NTOK, KVDIM, HIDDEN);

    gqa_flash<<<dim3(SEQ / 64, BATCH * N_HEADS), 128, 0, stream>>>(Qb, Kb, Vb, Cb);

    gemm_bf16<true><<<dim3(HIDDEN / 128, NTOK / 128), 256, 0, stream>>>(
        Cb, wob, bo, d_out, NTOK, HIDDEN, HIDDEN);
}
